// MoENetwork_83631603188335
// MI455X (gfx1250) — matrix operand order repaired and checked by reading, not compiled, not run on hardware
//
#include <hip/hip_runtime.h>
#include <hip/hip_bf16.h>
#include <cmath>

typedef _Float16 h16;
typedef __attribute__((ext_vector_type(8)))  _Float16 v8h;
typedef __attribute__((ext_vector_type(16))) _Float16 v16h;
typedef __attribute__((ext_vector_type(8)))  float    v8f;

#define B_   4096
#define E_   8
#define EPS_ 1e-5f

// ---------------- BatchNorm statistics (mean + rstd per column) ----------------
__global__ __launch_bounds__(256) void bn_stats_kernel(const float* __restrict__ x,
                                                       float* __restrict__ mu,
                                                       float* __restrict__ rstd,
                                                       int D) {
  const int col  = blockIdx.x * 32 + (threadIdx.x & 31);
  const int rgrp = threadIdx.x >> 5;  // 0..7
  float s = 0.f, q = 0.f;
  for (int r = rgrp; r < B_; r += 8) {
    float v = x[(size_t)r * D + col];
    s += v; q += v * v;
  }
  __shared__ float sh_s[8][32], sh_q[8][32];
  sh_s[rgrp][threadIdx.x & 31] = s;
  sh_q[rgrp][threadIdx.x & 31] = q;
  __syncthreads();
  if (rgrp == 0) {
    float ts = 0.f, tq = 0.f;
#pragma unroll
    for (int i = 0; i < 8; ++i) { ts += sh_s[i][threadIdx.x & 31]; tq += sh_q[i][threadIdx.x & 31]; }
    float m   = ts / (float)B_;
    float var = tq / (float)B_ - m * m;           // biased variance (matches reference)
    mu[col]   = m;
    rstd[col] = rsqrtf(var + EPS_);
  }
}

// ---------------- BatchNorm apply (+optional ReLU), write f16 ----------------
template <bool RELU>
__global__ __launch_bounds__(256) void bn_apply_kernel(const float* __restrict__ x,
                                                       const float* __restrict__ mu,
                                                       const float* __restrict__ rstd,
                                                       const float* __restrict__ gamma,
                                                       const float* __restrict__ beta,
                                                       h16* __restrict__ out,
                                                       int D, size_t total) {
  size_t i = (size_t)blockIdx.x * 256 + threadIdx.x;
  if (i >= total) return;
  int c = (int)(i % (size_t)D);
  float v = (x[i] - mu[c]) * rstd[c] * gamma[c] + beta[c];
  if (RELU) v = fmaxf(v, 0.f);
  out[i] = (h16)v;
}

// ---------------- Gating: logits, top-2 (with ties, ref semantics), softmax ----------------
template <int K>
__global__ __launch_bounds__(256) void gate_kernel(const h16* __restrict__ h,
                                                   const float* __restrict__ gW,  // [K, E]
                                                   const float* __restrict__ gb,  // [E]
                                                   float* __restrict__ gates) {   // [B, E]
  const int row  = blockIdx.x * 8 + (threadIdx.x >> 5);
  const int lane = threadIdx.x & 31;
  float acc[E_] = {};
  for (int k = lane; k < K; k += 32) {
    float hv = (float)h[(size_t)row * K + k];
    const float4* w = (const float4*)(gW + (size_t)k * E_);
    float4 w0 = w[0], w1 = w[1];
    acc[0] += hv * w0.x; acc[1] += hv * w0.y; acc[2] += hv * w0.z; acc[3] += hv * w0.w;
    acc[4] += hv * w1.x; acc[5] += hv * w1.y; acc[6] += hv * w1.z; acc[7] += hv * w1.w;
  }
#pragma unroll
  for (int off = 16; off > 0; off >>= 1)
#pragma unroll
    for (int e = 0; e < E_; ++e) acc[e] += __shfl_xor(acc[e], off, 32);
  if (lane == 0) {
    float l[E_];
#pragma unroll
    for (int e = 0; e < E_; ++e) l[e] = acc[e] + gb[e];
    int i1 = 0;
#pragma unroll
    for (int e = 1; e < E_; ++e) if (l[e] > l[i1]) i1 = e;
    float m1 = l[i1];
    float m2 = -INFINITY;
#pragma unroll
    for (int e = 0; e < E_; ++e) if (e != i1 && l[e] > m2) m2 = l[e];
    float z = 0.f, w[E_];
#pragma unroll
    for (int e = 0; e < E_; ++e) {
      w[e] = (l[e] >= m2) ? __expf(l[e] - m1) : 0.f;  // keeps ties, like reference
      z += w[e];
    }
    float inv = 1.f / z;
#pragma unroll
    for (int e = 0; e < E_; ++e) gates[(size_t)row * E_ + e] = w[e] * inv;
  }
}

// ---------------- Pack expert weights: [E,K,N] f32 -> [E][K/32][N][32] f16 ----------------
template <int K, int N>
__global__ __launch_bounds__(256) void pack_w_kernel(const float* __restrict__ W,
                                                     h16* __restrict__ Wp) {
  size_t i = (size_t)blockIdx.x * 256 + threadIdx.x;   // over E*K*N, coalesced read
  int n = (int)(i % (size_t)N);
  int k = (int)((i / (size_t)N) % (size_t)K);
  int e = (int)(i / ((size_t)N * K));
  size_t dst = ((((size_t)e * (K / 32) + (k >> 5)) * N + n) << 5) + (k & 31);
  Wp[dst] = (h16)W[i];
}

// ---------------- MoE GEMM with async global->LDS staging ----------------
// Block: 256 threads (8 waves) owns a 128x128 output tile. Double-buffered LDS
// stages A (128x32) and B_e (32x128, fragment-packed) per k-step via
// GLOBAL_LOAD_ASYNC_TO_LDS_B128 (ASYNCcnt). Gate folded into A per lane.
// Inner loop is branch-free: expert hand-off staging is peeled out of it.
template <int K, int N>
__global__ __launch_bounds__(256) void moe_gemm_kernel(const h16* __restrict__ A,   // [B,K]
                                                       const h16* __restrict__ Wp,  // packed
                                                       const float* __restrict__ gates,
                                                       float* __restrict__ Y) {     // [B,N]
  constexpr int KB = K / 32;
  constexpr int BN = N / 128;
  __shared__ __align__(64) h16 lsA[2][128 * 32];   // 8KB per buffer
  __shared__ __align__(64) h16 lsB[2][128 * 32];   // 8KB per buffer

  const int tid  = threadIdx.x;
  const int w    = tid >> 5;
  const int lane = tid & 31;
  const int half = lane >> 4, l16 = lane & 15;
  const int m0 = (blockIdx.x / BN) * 128;
  const int n0 = (blockIdx.x % BN) * 128;

  // wave tile within block: rows (w&3)*32..+31, cols (w>>2)*64..+63
  const int wr = (w & 3) * 32;
  const int wc = (w >> 2) * 64;
  const int row0 = wr + l16;        // local row, A frag mi=0
  const int row1 = wr + 16 + l16;   // mi=1

  // staging: 512 16B chunks each for A and B; thread handles chunk tid and tid+256
  const int cr0 = tid >> 2, cq0 = tid & 3;     // row/col 0..63, quarter 0..3
  const int cr1 = cr0 + 64, cq1 = cq0;

  v8f acc[2][4];
#pragma unroll
  for (int i = 0; i < 2; ++i)
#pragma unroll
    for (int j = 0; j < 4; ++j) acc[i][j] = (v8f){};

  auto stage = [&](int e, int kb, int buf) {
    // A chunks: lsA[row*32 + q*8] <- A[(m0+row)*K + kb*32 + q*8]
    {
      const h16* g0 = A + (size_t)(m0 + cr0) * K + kb * 32 + cq0 * 8;
      const h16* g1 = A + (size_t)(m0 + cr1) * K + kb * 32 + cq1 * 8;
      unsigned l0 = (unsigned)(size_t)(&lsA[buf][cr0 * 32 + cq0 * 8]);
      unsigned l1 = (unsigned)(size_t)(&lsA[buf][cr1 * 32 + cq1 * 8]);
      asm volatile("global_load_async_to_lds_b128 %0, %1, off" :: "v"(l0), "v"(g0) : "memory");
      asm volatile("global_load_async_to_lds_b128 %0, %1, off" :: "v"(l1), "v"(g1) : "memory");
    }
    // B chunks: lsB[col*32 + q*8] <- Wp[((e*KB+kb)*N + n0+col)*32 + q*8]
    {
      const h16* base = Wp + ((size_t)(e * KB + kb) * N + n0) * 32;
      const h16* g0 = base + (size_t)cr0 * 32 + cq0 * 8;
      const h16* g1 = base + (size_t)cr1 * 32 + cq1 * 8;
      unsigned l0 = (unsigned)(size_t)(&lsB[buf][cr0 * 32 + cq0 * 8]);
      unsigned l1 = (unsigned)(size_t)(&lsB[buf][cr1 * 32 + cq1 * 8]);
      asm volatile("global_load_async_to_lds_b128 %0, %1, off" :: "v"(l0), "v"(g0) : "memory");
      asm volatile("global_load_async_to_lds_b128 %0, %1, off" :: "v"(l1), "v"(g1) : "memory");
    }
  };

  // compute: fragment layouts are fixed LDS offsets (independent of kb/e)
  auto compute = [&](int buf, h16 g0, h16 g1) {
    const h16* LA = &lsA[buf][0];
    const h16* LB = &lsB[buf][0];
    v16h a0 = *(const v16h*)(LA + row0 * 32 + half * 16);
    v16h a1 = *(const v16h*)(LA + row1 * 32 + half * 16);
    a0 = a0 * g0;
    a1 = a1 * g1;
    v16h b0 = *(const v16h*)(LB + (wc + 0 * 16 + l16) * 32 + half * 16);
    v16h b1 = *(const v16h*)(LB + (wc + 1 * 16 + l16) * 32 + half * 16);
    v16h b2 = *(const v16h*)(LB + (wc + 2 * 16 + l16) * 32 + half * 16);
    v16h b3 = *(const v16h*)(LB + (wc + 3 * 16 + l16) * 32 + half * 16);
    acc[0][0] = __builtin_amdgcn_wmma_f32_16x16x32_f16(false, a0, false, b0, (short)0, acc[0][0], false, false);
    acc[1][0] = __builtin_amdgcn_wmma_f32_16x16x32_f16(false, a1, false, b0, (short)0, acc[1][0], false, false);
    acc[0][1] = __builtin_amdgcn_wmma_f32_16x16x32_f16(false, a0, false, b1, (short)0, acc[0][1], false, false);
    acc[1][1] = __builtin_amdgcn_wmma_f32_16x16x32_f16(false, a1, false, b1, (short)0, acc[1][1], false, false);
    acc[0][2] = __builtin_amdgcn_wmma_f32_16x16x32_f16(false, a0, false, b2, (short)0, acc[0][2], false, false);
    acc[1][2] = __builtin_amdgcn_wmma_f32_16x16x32_f16(false, a1, false, b2, (short)0, acc[1][2], false, false);
    acc[0][3] = __builtin_amdgcn_wmma_f32_16x16x32_f16(false, a0, false, b3, (short)0, acc[0][3], false, false);
    acc[1][3] = __builtin_amdgcn_wmma_f32_16x16x32_f16(false, a1, false, b3, (short)0, acc[1][3], false, false);
  };

  stage(0, 0, 0);
  asm volatile("s_wait_asynccnt 0x0" ::: "memory");
  __syncthreads();

  int buf = 0;
#pragma unroll 1
  for (int e = 0; e < E_; ++e) {
    const h16 g0 = (h16)gates[(size_t)(m0 + row0) * E_ + e];
    const h16 g1 = (h16)gates[(size_t)(m0 + row1) * E_ + e];
    // steady state: branch-free inner loop
#pragma unroll 1
    for (int kb = 0; kb < KB - 1; ++kb) {
      stage(e, kb + 1, buf ^ 1);
      compute(buf, g0, g1);
      asm volatile("s_wait_asynccnt 0x0" ::: "memory");
      __syncthreads();
      buf ^= 1;
    }
    // expert hand-off (peeled last k-step)
    if (e + 1 < E_) stage(e + 1, 0, buf ^ 1);
    compute(buf, g0, g1);
    asm volatile("s_wait_asynccnt 0x0" ::: "memory");
    __syncthreads();
    buf ^= 1;
  }

  // C/D layout: VGPR r, lanes 0-15 -> M=r, lanes 16-31 -> M=r+8; N = lane%16
#pragma unroll
  for (int ni = 0; ni < 4; ++ni) {
#pragma unroll
    for (int r = 0; r < 8; ++r) {
      int gr0 = m0 + wr + r + 8 * half;
      int gr1 = gr0 + 16;
      int gc  = n0 + wc + ni * 16 + l16;
      Y[(size_t)gr0 * N + gc] = acc[0][ni][r];
      Y[(size_t)gr1 * N + gc] = acc[1][ni][r];
    }
  }
}

// ---------------- Y += gates @ expert_bias ----------------
__global__ __launch_bounds__(256) void bias_mix_kernel(float* __restrict__ Y,
                                                       const float* __restrict__ gates,
                                                       const float* __restrict__ bias,  // [E,N]
                                                       int N, size_t total) {
  size_t i = (size_t)blockIdx.x * 256 + threadIdx.x;
  if (i >= total) return;
  int j    = (int)(i % (size_t)N);
  size_t b = i / (size_t)N;
  float s = 0.f;
#pragma unroll
  for (int e = 0; e < E_; ++e) s += gates[b * E_ + e] * bias[(size_t)e * N + j];
  Y[i] += s;
}

// ---------------- Head: out[b] = relu(y2[b,:]) . out_W + out_b ----------------
__global__ __launch_bounds__(256) void head_kernel(const float* __restrict__ Y,   // [B,1024]
                                                   const float* __restrict__ Wo,  // [1024]
                                                   const float* __restrict__ bo,  // [1]
                                                   float* __restrict__ out) {
  const int row  = blockIdx.x * 8 + (threadIdx.x >> 5);
  const int lane = threadIdx.x & 31;
  float s = 0.f;
  for (int k = lane; k < 1024; k += 32) {
    float v = fmaxf(Y[(size_t)row * 1024 + k], 0.f);
    s += v * Wo[k];
  }
#pragma unroll
  for (int off = 16; off > 0; off >>= 1) s += __shfl_xor(s, off, 32);
  if (lane == 0) out[row] = s + bo[0];
}

extern "C" void kernel_launch(void* const* d_in, const int* in_sizes, int n_in,
                              void* d_out, int out_size, void* d_ws, size_t ws_size,
                              hipStream_t stream) {
  (void)in_sizes; (void)n_in; (void)out_size; (void)ws_size;
  const float* x     = (const float*)d_in[0];
  const float* bn1_g = (const float*)d_in[1];
  const float* bn1_b = (const float*)d_in[2];
  const float* bn2_g = (const float*)d_in[3];
  const float* bn2_b = (const float*)d_in[4];
  const float* g1W   = (const float*)d_in[5];
  const float* g1b   = (const float*)d_in[6];
  const float* e1W   = (const float*)d_in[7];
  const float* e1b   = (const float*)d_in[8];
  const float* g2W   = (const float*)d_in[9];
  const float* g2b   = (const float*)d_in[10];
  const float* e2W   = (const float*)d_in[11];
  const float* e2b   = (const float*)d_in[12];
  const float* oW    = (const float*)d_in[13];
  const float* ob    = (const float*)d_in[14];
  float* out = (float*)d_out;

  char* p = (char*)d_ws;
  auto alloc = [&](size_t bytes) -> char* {
    char* r = p; p += (bytes + 255) & ~(size_t)255; return r;
  };
  float* mu1 = (float*)alloc(1024 * 4);
  float* rs1 = (float*)alloc(1024 * 4);
  float* mu2 = (float*)alloc(2048 * 4);
  float* rs2 = (float*)alloc(2048 * 4);
  h16*   h1  = (h16*)  alloc((size_t)4096 * 1024 * 2);
  h16*   h2  = (h16*)  alloc((size_t)4096 * 2048 * 2);
  float* gt1 = (float*)alloc((size_t)4096 * 8 * 4);
  float* gt2 = (float*)alloc((size_t)4096 * 8 * 4);
  h16*   Wp1 = (h16*)  alloc((size_t)8 * 1024 * 2048 * 2);
  h16*   Wp2 = (h16*)  alloc((size_t)8 * 2048 * 1024 * 2);
  float* y1  = (float*)alloc((size_t)4096 * 2048 * 4);
  float* y2  = (float*)alloc((size_t)4096 * 1024 * 4);

  // ---- layer 1: BN -> gate -> MoE GEMM -> bias mix ----
  bn_stats_kernel<<<1024 / 32, 256, 0, stream>>>(x, mu1, rs1, 1024);
  bn_apply_kernel<false><<<(4096u * 1024u) / 256, 256, 0, stream>>>(
      x, mu1, rs1, bn1_g, bn1_b, h1, 1024, (size_t)4096 * 1024);
  gate_kernel<1024><<<4096 / 8, 256, 0, stream>>>(h1, g1W, g1b, gt1);
  pack_w_kernel<1024, 2048><<<(8u * 1024u * 2048u) / 256, 256, 0, stream>>>(e1W, Wp1);
  moe_gemm_kernel<1024, 2048><<<(4096 / 128) * (2048 / 128), 256, 0, stream>>>(h1, Wp1, gt1, y1);
  bias_mix_kernel<<<(4096u * 2048u) / 256, 256, 0, stream>>>(y1, gt1, e1b, 2048, (size_t)4096 * 2048);

  // ---- layer 2: BN+ReLU -> gate -> MoE GEMM -> bias mix ----
  bn_stats_kernel<<<2048 / 32, 256, 0, stream>>>(y1, mu2, rs2, 2048);
  bn_apply_kernel<true><<<(4096u * 2048u) / 256, 256, 0, stream>>>(
      y1, mu2, rs2, bn2_g, bn2_b, h2, 2048, (size_t)4096 * 2048);
  gate_kernel<2048><<<4096 / 8, 256, 0, stream>>>(h2, g2W, g2b, gt2);
  pack_w_kernel<2048, 1024><<<(8u * 2048u * 1024u) / 256, 256, 0, stream>>>(e2W, Wp2);
  moe_gemm_kernel<2048, 1024><<<(4096 / 128) * (1024 / 128), 256, 0, stream>>>(h2, Wp2, gt2, y2);
  bias_mix_kernel<<<(4096u * 1024u) / 256, 256, 0, stream>>>(y2, gt2, e2b, 1024, (size_t)4096 * 1024);

  // ---- head: relu + dot ----
  head_kernel<<<4096 / 8, 256, 0, stream>>>(y2, oW, ob, out);
}
